// ConcatTetraPermuter_32469952758017
// MI455X (gfx1250) — matrix operand order repaired and checked by reading, not compiled, not run on hardware
//
#include <hip/hip_runtime.h>

typedef unsigned short u16;
typedef unsigned int   u32;
typedef __attribute__((ext_vector_type(8)))  __bf16 v8bf;
typedef __attribute__((ext_vector_type(16))) __bf16 v16bf;
typedef __attribute__((ext_vector_type(8)))  float  v8f;

#define HID 512
#define BN_EPS 1e-5f

__device__ __forceinline__ u16 f2bf(float f) {
    u32 u = __builtin_bit_cast(u32, f);
    u32 r = u + 0x7FFFu + ((u >> 16) & 1u);   // round-to-nearest-even
    return (u16)(r >> 16);
}
__device__ __forceinline__ float bf2f(u16 h) {
    u32 u = ((u32)h) << 16;
    return __builtin_bit_cast(float, u);
}
__device__ __forceinline__ u32 pack2(float a, float b) {
    return (u32)f2bf(a) | ((u32)f2bf(b) << 16);
}

// ---------------------------------------------------------------------------
// f32 -> bf16 elementwise convert, 4 elements per thread
// ---------------------------------------------------------------------------
__global__ __launch_bounds__(256) void k_f32_to_bf16(const float* __restrict__ src,
                                                     u16* __restrict__ dst, size_t n4) {
    size_t idx = (size_t)blockIdx.x * blockDim.x + threadIdx.x;
    if (idx >= n4) return;
    float4 v = *(const float4*)(src + idx * 4);
    uint2 o;
    o.x = pack2(v.x, v.y);
    o.y = pack2(v.z, v.w);
    *(uint2*)(dst + idx * 4) = o;
}

// ---------------------------------------------------------------------------
// WbigT[c][i] = W_bs[c%512, (c/512)*512 + i]  (bf16, Bt layout: Nc x K row-major)
// ---------------------------------------------------------------------------
__global__ __launch_bounds__(256) void k_prep_wbig(const float* __restrict__ W_bs,
                                                   u16* __restrict__ WbigT) {
    int idx = blockIdx.x * blockDim.x + threadIdx.x;   // 2048*512 total
    if (idx >= 2048 * 512) return;
    int c = idx >> 9;
    int i = idx & 511;
    int s = c >> 9;
    int kc = c & 511;
    WbigT[idx] = f2bf(W_bs[(size_t)kc * 2048 + s * 512 + i]);
}

// ---------------------------------------------------------------------------
// Fused GEMM1 + 12-perm relu-sum.
// Block: 16 n (=64 Xflat rows) x 64 output features x all 4 s-chunks.
// Accumulate z[64 rows][4s x 64f] with WMMA, exchange via LDS, then
// m[n,f] = (1/3) * sum_p relu( sum_s z[4n+p[s]][s*64+f] + b[f] ).
// ---------------------------------------------------------------------------
__global__ __launch_bounds__(256) void k_gemm_permsum(
    const u16* __restrict__ Xbf,    // (4N x 512) bf16
    const u16* __restrict__ WbigT,  // (2048 x 512) bf16, row c = s*512+f
    const float* __restrict__ b_bs, // (512,)
    u16* __restrict__ m_out,        // (N x 512) bf16
    int N)
{
    // LDS: staging (As 64x72 + Bs 256x72 bf16 = 45KB) aliased with E (64x256 f32 = 64KB)
    __shared__ __align__(16) char smraw[64 * 256 * 4];
    u16*   As = (u16*)smraw;                    // 64*72  u16 = 9216 B
    u16*   Bs = (u16*)(smraw + 64 * 72 * 2);    // 256*72 u16 = 36864 B (16B-aligned)
    float* E  = (float*)smraw;                  // 64 x 256 f32

    const int tid  = threadIdx.x;
    const int lane = tid & 31;
    const int wave = tid >> 5;
    const int waveM = wave & 1;    // 32-row half
    const int sQ    = wave >> 1;   // s-chunk / 64-col quarter
    const int lrow = lane & 15;
    const int lhi  = lane >> 4;

    const int nb = blockIdx.y;          // n-block (16 n per block)
    const int f0 = blockIdx.x * 64;     // feature block
    const int M1 = 4 * N;

    const int chunk = tid & 7;          // 16B chunk within 64-wide K slice
    const int rbase = tid >> 3;         // 0..31

    v8f acc[2][4];
#pragma unroll
    for (int mi = 0; mi < 2; ++mi)
#pragma unroll
        for (int ni = 0; ni < 4; ++ni)
            acc[mi][ni] = (v8f){0.f, 0.f, 0.f, 0.f, 0.f, 0.f, 0.f, 0.f};

    for (int k0 = 0; k0 < HID; k0 += 64) {
        // ---- stage A tile: Xflat rows [64*nb, 64*nb+64), 64 K ----
#pragma unroll
        for (int i = 0; i < 2; ++i) {
            int row = rbase + 32 * i;
            int gm = nb * 64 + row;
            uint4 v = {0u, 0u, 0u, 0u};
            if (gm < M1) {
                const u16* gp = Xbf + (size_t)gm * HID + k0 + chunk * 8;
                v = *(const uint4*)gp;
                if (k0 + 64 < HID) __builtin_prefetch(gp + 64, 0, 1);
            }
            *(uint4*)(As + row * 72 + chunk * 8) = v;
        }
        // ---- stage B tile: WbigT rows {s*512 + f0 + fl}, 256 rows x 64 K ----
#pragma unroll
        for (int i = 0; i < 8; ++i) {
            int row = rbase + 32 * i;              // 0..255
            int s  = row >> 6;
            int fl = row & 63;
            const u16* gp = WbigT + (size_t)(s * 512 + f0 + fl) * HID + k0 + chunk * 8;
            uint4 v = *(const uint4*)gp;
            if (k0 + 64 < HID) __builtin_prefetch(gp + 64, 0, 1);
            *(uint4*)(Bs + row * 72 + chunk * 8) = v;
        }
        __syncthreads();

#pragma unroll
        for (int kk = 0; kk < 64; kk += 32) {
            v16bf afr[2];
            v16bf bfr[4];
#pragma unroll
            for (int mi = 0; mi < 2; ++mi) {
                const u16* p = As + (waveM * 32 + mi * 16 + lrow) * 72 + kk;
                v8bf lo = *(const v8bf*)(p + lhi * 8);
                v8bf hi = *(const v8bf*)(p + 16 + lhi * 8);
                afr[mi] = __builtin_shufflevector(lo, hi, 0,1,2,3,4,5,6,7,8,9,10,11,12,13,14,15);
            }
#pragma unroll
            for (int ni = 0; ni < 4; ++ni) {
                const u16* p = Bs + (sQ * 64 + ni * 16 + lrow) * 72 + kk;
                v8bf lo = *(const v8bf*)(p + lhi * 8);
                v8bf hi = *(const v8bf*)(p + 16 + lhi * 8);
                bfr[ni] = __builtin_shufflevector(lo, hi, 0,1,2,3,4,5,6,7,8,9,10,11,12,13,14,15);
            }
#pragma unroll
            for (int mi = 0; mi < 2; ++mi)
#pragma unroll
                for (int ni = 0; ni < 4; ++ni)
                    acc[mi][ni] = __builtin_amdgcn_wmma_f32_16x16x32_bf16(
                        false, afr[mi], false, bfr[ni], (short)0, acc[mi][ni], false, false);
        }
        __syncthreads();
    }

    // ---- dump accumulators to LDS exchange buffer (aliases staging; safe after sync) ----
#pragma unroll
    for (int mi = 0; mi < 2; ++mi) {
#pragma unroll
        for (int ni = 0; ni < 4; ++ni) {
            int col = sQ * 64 + ni * 16 + lrow;
#pragma unroll
            for (int v = 0; v < 8; ++v) {
                int row = waveM * 32 + mi * 16 + v + lhi * 8;
                E[row * 256 + col] = acc[mi][ni][v];
            }
        }
    }
    __syncthreads();

    // ---- 12-perm relu-sum from LDS ----
    const int P[12][4] = {
        {0,1,2,3},{0,2,3,1},{0,3,1,2},{1,0,3,2},{1,2,0,3},{1,3,2,0},
        {2,0,1,3},{2,1,3,0},{2,3,0,1},{3,0,2,1},{3,1,0,2},{3,2,1,0}};

    const int kl = tid & 63;
    const float bias = b_bs[f0 + kl];
    const float inv3 = 1.0f / 3.0f;
#pragma unroll
    for (int i = 0; i < 4; ++i) {
        int nl = (tid >> 6) + 4 * i;            // 0..15
        int n = nb * 16 + nl;
        if (n >= N) continue;
        float zz[4][4];
#pragma unroll
        for (int j = 0; j < 4; ++j)
#pragma unroll
            for (int s = 0; s < 4; ++s)
                zz[j][s] = E[(4 * nl + j) * 256 + s * 64 + kl];
        float a = 0.f;
#pragma unroll
        for (int p = 0; p < 12; ++p) {
            float v = zz[P[p][0]][0] + zz[P[p][1]][1] +
                      zz[P[p][2]][2] + zz[P[p][3]][3] + bias;
            a += fmaxf(v, 0.f);
        }
        m_out[(size_t)n * HID + f0 + kl] = f2bf(a * inv3);
    }
}

// ---------------------------------------------------------------------------
// Tiled bf16 WMMA GEMM:  C[M x Nc] = A[M x K] * Bt^T   (Bt is Nc x K row-major)
// 128x128x64 block tile, 8 waves, each wave owns 64x32 via 4x2 grid of 16x16.
// ---------------------------------------------------------------------------
template <bool HAS_BIAS, bool OUT_BF16>
__global__ __launch_bounds__(256) void gemm_bf16_wmma(
    const u16* __restrict__ A, const u16* __restrict__ Bt,
    const float* __restrict__ bias, void* __restrict__ Cout,
    int M, int Nc, int K)
{
    __shared__ u16 As[128 * 72];
    __shared__ u16 Bs[128 * 72];

    const int tid   = threadIdx.x;
    const int lane  = tid & 31;
    const int wave  = tid >> 5;
    const int waveM = wave & 1;
    const int waveN = wave >> 1;
    const int m0 = blockIdx.y * 128;
    const int n0 = blockIdx.x * 128;

    const int lrow = lane & 15;
    const int lhi  = lane >> 4;

    const int chunk = tid & 7;
    const int rbase = tid >> 3;

    v8f acc[4][2];
#pragma unroll
    for (int mi = 0; mi < 4; ++mi)
#pragma unroll
        for (int ni = 0; ni < 2; ++ni)
            acc[mi][ni] = (v8f){0.f, 0.f, 0.f, 0.f, 0.f, 0.f, 0.f, 0.f};

    for (int k0 = 0; k0 < K; k0 += 64) {
#pragma unroll
        for (int i = 0; i < 4; ++i) {
            int row = rbase + 32 * i;
            int gm = m0 + row;
            uint4 v = {0u, 0u, 0u, 0u};
            if (gm < M) {
                const u16* gp = A + (size_t)gm * K + k0 + chunk * 8;
                v = *(const uint4*)gp;
                if (k0 + 64 < K) __builtin_prefetch(gp + 64, 0, 1);
            }
            *(uint4*)(As + row * 72 + chunk * 8) = v;
        }
#pragma unroll
        for (int i = 0; i < 4; ++i) {
            int col = rbase + 32 * i;
            int gn = n0 + col;
            uint4 v = {0u, 0u, 0u, 0u};
            if (gn < Nc) {
                const u16* gp = Bt + (size_t)gn * K + k0 + chunk * 8;
                v = *(const uint4*)gp;
                if (k0 + 64 < K) __builtin_prefetch(gp + 64, 0, 1);
            }
            *(uint4*)(Bs + col * 72 + chunk * 8) = v;
        }
        __syncthreads();

#pragma unroll
        for (int kk = 0; kk < 64; kk += 32) {
            v16bf afr[4];
            v16bf bfr[2];
#pragma unroll
            for (int mi = 0; mi < 4; ++mi) {
                const u16* p = As + (waveM * 64 + mi * 16 + lrow) * 72 + kk;
                v8bf lo = *(const v8bf*)(p + lhi * 8);
                v8bf hi = *(const v8bf*)(p + 16 + lhi * 8);
                afr[mi] = __builtin_shufflevector(lo, hi, 0,1,2,3,4,5,6,7,8,9,10,11,12,13,14,15);
            }
#pragma unroll
            for (int ni = 0; ni < 2; ++ni) {
                const u16* p = Bs + (waveN * 32 + ni * 16 + lrow) * 72 + kk;
                v8bf lo = *(const v8bf*)(p + lhi * 8);
                v8bf hi = *(const v8bf*)(p + 16 + lhi * 8);
                bfr[ni] = __builtin_shufflevector(lo, hi, 0,1,2,3,4,5,6,7,8,9,10,11,12,13,14,15);
            }
#pragma unroll
            for (int mi = 0; mi < 4; ++mi)
#pragma unroll
                for (int ni = 0; ni < 2; ++ni)
                    acc[mi][ni] = __builtin_amdgcn_wmma_f32_16x16x32_bf16(
                        false, afr[mi], false, bfr[ni], (short)0, acc[mi][ni], false, false);
        }
        __syncthreads();
    }

#pragma unroll
    for (int mi = 0; mi < 4; ++mi) {
#pragma unroll
        for (int ni = 0; ni < 2; ++ni) {
            int gcol = n0 + waveN * 32 + ni * 16 + lrow;
            float bv = 0.0f;
            if constexpr (HAS_BIAS) bv = bias[gcol];
#pragma unroll
            for (int v = 0; v < 8; ++v) {
                int grow = m0 + waveM * 64 + mi * 16 + v + lhi * 8;
                if (grow < M) {
                    float val = acc[mi][ni][v] + bv;
                    if constexpr (OUT_BF16)
                        ((u16*)Cout)[(size_t)grow * Nc + gcol] = f2bf(val);
                    else
                        ((float*)Cout)[(size_t)grow * Nc + gcol] = val;
                }
            }
        }
    }
}

// ---------------------------------------------------------------------------
// Batch-norm stats over axis 0: stats[c] = mean, stats[512+c] = rsqrt(var+eps)
// ---------------------------------------------------------------------------
__global__ __launch_bounds__(256) void k_bnstats(const float* __restrict__ h1,
                                                 float* __restrict__ stats, int N)
{
    __shared__ float sh1[256];
    __shared__ float sh2[256];
    int tid = threadIdx.x;
    int c = blockIdx.x * 64 + (tid & 63);
    int g = tid >> 6;
    float s1 = 0.f, s2 = 0.f;
    for (int r = g; r < N; r += 4) {
        float v = h1[(size_t)r * 512 + c];
        s1 += v;
        s2 += v * v;
    }
    sh1[tid] = s1;
    sh2[tid] = s2;
    __syncthreads();
    if (g == 0) {
        float t1 = sh1[tid] + sh1[tid + 64] + sh1[tid + 128] + sh1[tid + 192];
        float t2 = sh2[tid] + sh2[tid + 64] + sh2[tid + 128] + sh2[tid + 192];
        float invN = 1.0f / (float)N;
        float mean = t1 * invN;
        float var = t2 * invN - mean * mean;
        stats[c] = mean;
        stats[512 + c] = rsqrtf(var + BN_EPS);
    }
}

// ---------------------------------------------------------------------------
// a2[n,c] = bf16( relu( (h1 - mean) * rstd * gamma + beta ) ), 4 per thread
// ---------------------------------------------------------------------------
__global__ __launch_bounds__(256) void k_normalize(const float* __restrict__ h1,
                                                   const float* __restrict__ stats,
                                                   const float* __restrict__ gamma,
                                                   const float* __restrict__ beta,
                                                   u16* __restrict__ a2, size_t n4)
{
    size_t idx = (size_t)blockIdx.x * blockDim.x + threadIdx.x;
    if (idx >= n4) return;
    size_t e0 = idx * 4;
    int c0 = (int)(e0 & 511);
    float4 h = *(const float4*)(h1 + e0);
    float4 mn = *(const float4*)(stats + c0);
    float4 rs = *(const float4*)(stats + 512 + c0);
    float4 gm = *(const float4*)(gamma + c0);
    float4 bt = *(const float4*)(beta + c0);
    float r0 = fmaxf((h.x - mn.x) * rs.x * gm.x + bt.x, 0.f);
    float r1 = fmaxf((h.y - mn.y) * rs.y * gm.y + bt.y, 0.f);
    float r2 = fmaxf((h.z - mn.z) * rs.z * gm.z + bt.z, 0.f);
    float r3 = fmaxf((h.w - mn.w) * rs.w * gm.w + bt.w, 0.f);
    uint2 o;
    o.x = pack2(r0, r1);
    o.y = pack2(r2, r3);
    *(uint2*)(a2 + e0) = o;
}

// ---------------------------------------------------------------------------
extern "C" void kernel_launch(void* const* d_in, const int* in_sizes, int n_in,
                              void* d_out, int out_size, void* d_ws, size_t ws_size,
                              hipStream_t stream)
{
    const float* x     = (const float*)d_in[0];  // (N,4,512)
    const float* W_bs  = (const float*)d_in[1];  // (512,2048)
    const float* b_bs  = (const float*)d_in[2];  // (512,)
    const float* w1    = (const float*)d_in[3];  // (512,512)
    const float* b1    = (const float*)d_in[4];
    const float* gamma = (const float*)d_in[5];
    const float* beta  = (const float*)d_in[6];
    const float* w2    = (const float*)d_in[7];  // (512,512)
    const float* b2    = (const float*)d_in[8];

    const int N = in_sizes[0] / (4 * HID);   // 50000
    const int M1 = 4 * N;

    // ---- workspace carve-up (~410 MB total) ----
    char* ws = (char*)d_ws;
    size_t off = 0;
    auto carve = [&](size_t bytes) -> void* {
        void* p = ws + off;
        off = (off + bytes + 255) & ~(size_t)255;
        return p;
    };
    u16*   Xbf   = (u16*)carve((size_t)M1 * HID * 2);          // 204.8 MB
    u16*   WbigT = (u16*)carve((size_t)2048 * HID * 2);        // 2 MB
    u16*   w1bf  = (u16*)carve((size_t)HID * HID * 2);
    u16*   w2bf  = (u16*)carve((size_t)HID * HID * 2);
    u16*   Mbf   = (u16*)carve((size_t)N * HID * 2);           // 51.2 MB
    float* H1    = (float*)carve((size_t)N * HID * 4);         // 102.4 MB
    float* Stats = (float*)carve((size_t)1024 * 4);
    u16*   A2    = (u16*)carve((size_t)N * HID * 2);           // 51.2 MB
    (void)ws_size;

    // 1) bf16 conversions
    {
        size_t n4 = (size_t)M1 * HID / 4;
        k_f32_to_bf16<<<(unsigned)((n4 + 255) / 256), 256, 0, stream>>>(x, Xbf, n4);
    }
    k_prep_wbig<<<(2048 * 512 + 255) / 256, 256, 0, stream>>>(W_bs, WbigT);
    {
        size_t n4 = (size_t)HID * HID / 4;
        k_f32_to_bf16<<<(unsigned)((n4 + 255) / 256), 256, 0, stream>>>(w1, w1bf, n4);
        k_f32_to_bf16<<<(unsigned)((n4 + 255) / 256), 256, 0, stream>>>(w2, w2bf, n4);
    }

    // 2) fused: m = permsum( Xflat @ Wbig ) / 3   (no Z intermediate)
    {
        dim3 grid(HID / 64, (N + 15) / 16);
        k_gemm_permsum<<<grid, 256, 0, stream>>>(Xbf, WbigT, b_bs, Mbf, N);
    }

    // 3) h1 = m @ w1^T + b1  (Bt = w1 row-major directly), f32 out
    {
        dim3 grid(HID / 128, (N + 127) / 128);
        gemm_bf16_wmma<true, false><<<grid, 256, 0, stream>>>(
            Mbf, w1bf, b1, (void*)H1, N, HID, HID);
    }

    // 4) batch-norm stats
    k_bnstats<<<8, 256, 0, stream>>>(H1, Stats, N);

    // 5) normalize + relu -> bf16
    {
        size_t n4 = (size_t)N * HID / 4;
        k_normalize<<<(unsigned)((n4 + 255) / 256), 256, 0, stream>>>(
            H1, Stats, gamma, beta, A2, n4);
    }

    // 6) out = a2 @ w2^T + b2  (f32 into d_out)
    {
        dim3 grid(HID / 128, (N + 127) / 128);
        gemm_bf16_wmma<true, false><<<grid, 256, 0, stream>>>(
            A2, w2bf, b2, d_out, N, HID, HID);
    }
}
